// GCNModel_75651553951873
// MI455X (gfx1250) — 16-bit and 8-bit matrix operand order repaired and checked by reading, compiled, not run on hardware
//
#include <hip/hip_runtime.h>
#include <hip/hip_bf16.h>
#include <cstdint>


#define NN   4096
#define NFD  512
#define NHD  128
#define NCD  40

typedef __attribute__((ext_vector_type(16))) _Float16 v16h_t;
typedef __attribute__((ext_vector_type(8)))  _Float16 v8h_t;
typedef __attribute__((ext_vector_type(4)))  _Float16 v4h_t;
typedef __attribute__((ext_vector_type(8)))  float    v8f_t;
typedef __attribute__((ext_vector_type(4)))  float    f32x4_t;
typedef __attribute__((ext_vector_type(8)))  int      v8i_t;
typedef __attribute__((ext_vector_type(4)))  unsigned int u32x4_t;
typedef __attribute__((ext_vector_type(2)))  unsigned int u32x2_t;

template<int TRANSB>
__global__ __launch_bounds__(256)
void gemm_f16_wmma(const float* __restrict__ A, int lda,
                   const float* __restrict__ B, int ldb,
                   float* __restrict__ C, int ldc,
                   int M, int N, int K, float alpha, int betaFlag)
{
    __shared__ _Float16 sA[64][72];
    __shared__ _Float16 sB[64][72];

    const int tid  = threadIdx.x;
    const int wave = tid >> 5;
    const int lane = tid & 31;
    const int hl   = lane >> 4;
    const int l16  = lane & 15;

    const int m0 = blockIdx.y * 64;
    const int n0 = blockIdx.x * 64;
    const int wm = (wave & 3) * 16;
    const int wn = (wave >> 2) * 32;

    const bool interior = (m0 + 64 <= M) && (n0 + 64 <= N);

    v8f_t acc0 = {};
    v8f_t acc1 = {};

    const int nk = (K + 63) >> 6;
    for (int kt = 0; kt < nk; ++kt) {
        const int k0 = kt << 6;

        if (interior && (k0 + 64 <= K)) {
            f32x4_t av[4], bv[4];
            #pragma unroll
            for (int i = 0; i < 4; ++i) {
                int s = tid + (i << 8);
                int r = s >> 4, c4 = (s & 15) << 2;
                av[i] = *(const f32x4_t*)&A[(size_t)(m0 + r) * lda + k0 + c4];
            }
            if (TRANSB) {
                #pragma unroll
                for (int i = 0; i < 4; ++i) {
                    int s = tid + (i << 8);
                    int r = s >> 4, c4 = (s & 15) << 2;
                    bv[i] = *(const f32x4_t*)&B[(size_t)(n0 + r) * ldb + k0 + c4];
                }
            } else {
                #pragma unroll
                for (int i = 0; i < 4; ++i) {
                    int s = tid + (i << 8);
                    int k = s >> 4, n4 = (s & 15) << 2;
                    bv[i] = *(const f32x4_t*)&B[(size_t)(k0 + k) * ldb + n0 + n4];
                }
            }
            {
                int kp = (kt + 1) << 6;
                if (kp + 64 <= K)
                    __builtin_prefetch(&A[(size_t)(m0 + (tid >> 2)) * lda + kp +
                                          ((tid & 3) << 4)], 0, 1);
            }
            #pragma unroll
            for (int i = 0; i < 4; ++i) {
                int s = tid + (i << 8);
                int r = s >> 4, c4 = (s & 15) << 2;
                v4h_t h = { (_Float16)av[i][0], (_Float16)av[i][1],
                            (_Float16)av[i][2], (_Float16)av[i][3] };
                *(v4h_t*)&sA[r][c4] = h;
            }
            if (TRANSB) {
                #pragma unroll
                for (int i = 0; i < 4; ++i) {
                    int s = tid + (i << 8);
                    int r = s >> 4, c4 = (s & 15) << 2;
                    v4h_t h = { (_Float16)bv[i][0], (_Float16)bv[i][1],
                                (_Float16)bv[i][2], (_Float16)bv[i][3] };
                    *(v4h_t*)&sB[r][c4] = h;
                }
            } else {
                #pragma unroll
                for (int i = 0; i < 4; ++i) {
                    int s = tid + (i << 8);
                    int k = s >> 4, n4 = (s & 15) << 2;
                    sB[n4 + 0][k] = (_Float16)bv[i][0];
                    sB[n4 + 1][k] = (_Float16)bv[i][1];
                    sB[n4 + 2][k] = (_Float16)bv[i][2];
                    sB[n4 + 3][k] = (_Float16)bv[i][3];
                }
            }
        } else {
            #pragma unroll
            for (int i = 0; i < 16; ++i) {
                int s = tid + (i << 8);
                int r = s >> 6, c = s & 63;
                int gr = m0 + r, gk = k0 + c;
                bool ok = (gr < M) && (gk < K);
                float v = A[ok ? ((size_t)gr * lda + gk) : (size_t)0];
                sA[r][c] = (_Float16)(ok ? v : 0.0f);
            }
            if (TRANSB) {
                #pragma unroll
                for (int i = 0; i < 16; ++i) {
                    int s = tid + (i << 8);
                    int n = s >> 6, c = s & 63;
                    int gn = n0 + n, gk = k0 + c;
                    bool ok = (gn < N) && (gk < K);
                    float v = B[ok ? ((size_t)gn * ldb + gk) : (size_t)0];
                    sB[n][c] = (_Float16)(ok ? v : 0.0f);
                }
            } else {
                #pragma unroll
                for (int i = 0; i < 16; ++i) {
                    int s = tid + (i << 8);
                    int k = s >> 6, n = s & 63;
                    int gn = n0 + n, gk = k0 + k;
                    bool ok = (gn < N) && (gk < K);
                    float v = B[ok ? ((size_t)gk * ldb + gn) : (size_t)0];
                    sB[n][k] = (_Float16)(ok ? v : 0.0f);
                }
            }
        }
        __syncthreads();

        #pragma unroll
        for (int kk = 0; kk < 64; kk += 32) {
            union { v16h_t v; v8h_t h[2]; } fa, fb0, fb1;
            const int am = wm + l16;
            fa.h[0] = *(const v8h_t*)&sA[am][kk + hl * 16];
            fa.h[1] = *(const v8h_t*)&sA[am][kk + hl * 16 + 8];
            const int bn0 = wn + l16;
            const int bn1 = bn0 + 16;
            fb0.h[0] = *(const v8h_t*)&sB[bn0][kk + hl * 16];
            fb0.h[1] = *(const v8h_t*)&sB[bn0][kk + hl * 16 + 8];
            fb1.h[0] = *(const v8h_t*)&sB[bn1][kk + hl * 16];
            fb1.h[1] = *(const v8h_t*)&sB[bn1][kk + hl * 16 + 8];

            acc0 = __builtin_amdgcn_wmma_f32_16x16x32_f16(false, fa.v, false, fb0.v,
                                                          (short)0, acc0, false, false);
            acc1 = __builtin_amdgcn_wmma_f32_16x16x32_f16(false, fa.v, false, fb1.v,
                                                          (short)0, acc1, false, false);
        }
        __syncthreads();
    }

    #pragma unroll
    for (int r = 0; r < 8; ++r) {
        int row = m0 + wm + hl * 8 + r;
        if (row < M) {
            int col0 = n0 + wn + l16;
            int col1 = col0 + 16;
            if (col0 < N) {
                size_t o = (size_t)row * ldc + col0;
                C[o] = alpha * acc0[r] + (betaFlag ? C[o] : 0.0f);
            }
            if (col1 < N) {
                size_t o = (size_t)row * ldc + col1;
                C[o] = alpha * acc1[r] + (betaFlag ? C[o] : 0.0f);
            }
        }
    }
}

__global__ __launch_bounds__(256)
void adj2_mask_wmma(const unsigned char* __restrict__ F,
                    unsigned char* __restrict__ Mout, int N)
{
    __shared__ unsigned char sA[64][80];
    __shared__ unsigned char sB[64][80];

    const int tid  = threadIdx.x;
    const int wave = tid >> 5;
    const int lane = tid & 31;
    const int hl   = lane >> 4;
    const int l16  = lane & 15;

    const int m0 = blockIdx.y * 64;
    const int n0 = blockIdx.x * 64;
    const int wm = (wave & 3) * 16;
    const int wn = (wave >> 2) * 32;

    v8i_t acc0 = {};
    v8i_t acc1 = {};

    const int nk = N >> 6;
    for (int kt = 0; kt < nk; ++kt) {
        const int k0 = kt << 6;
        {
            int r  = tid >> 2;
            int cb = (tid & 3) << 4;
            u32x4_t va = *(const u32x4_t*)&F[(size_t)(m0 + r) * N + k0 + cb];
            union { u32x4_t v; unsigned char b[16]; } t;
            t.v = *(const u32x4_t*)&F[(size_t)(k0 + r) * N + n0 + cb];
            *(u32x4_t*)&sA[r][cb] = va;
            #pragma unroll
            for (int i = 0; i < 16; ++i) sB[cb + i][r] = (unsigned char)t.b[i];
        }
        __syncthreads();

        const int am = wm + l16;
        u32x4_t a0 = *(const u32x4_t*)&sA[am][hl * 16];
        u32x4_t a1 = *(const u32x4_t*)&sA[am][hl * 16 + 32];
        v8i_t fa = { (int)a0[0], (int)a0[1], (int)a0[2], (int)a0[3],
                     (int)a1[0], (int)a1[1], (int)a1[2], (int)a1[3] };

        const int bn0 = wn + l16;
        const int bn1 = bn0 + 16;
        u32x4_t b0a = *(const u32x4_t*)&sB[bn0][hl * 16];
        u32x4_t b0b = *(const u32x4_t*)&sB[bn0][hl * 16 + 32];
        v8i_t fb0 = { (int)b0a[0], (int)b0a[1], (int)b0a[2], (int)b0a[3],
                      (int)b0b[0], (int)b0b[1], (int)b0b[2], (int)b0b[3] };
        u32x4_t b1a = *(const u32x4_t*)&sB[bn1][hl * 16];
        u32x4_t b1b = *(const u32x4_t*)&sB[bn1][hl * 16 + 32];
        v8i_t fb1 = { (int)b1a[0], (int)b1a[1], (int)b1a[2], (int)b1a[3],
                      (int)b1b[0], (int)b1b[1], (int)b1b[2], (int)b1b[3] };

        acc0 = __builtin_amdgcn_wmma_i32_16x16x64_iu8(false, fa, false, fb0,
                                                      acc0, false, false);
        acc1 = __builtin_amdgcn_wmma_i32_16x16x64_iu8(false, fa, false, fb1,
                                                      acc1, false, false);
        __syncthreads();
    }

    #pragma unroll
    for (int r = 0; r < 8; ++r) {
        int row  = m0 + wm + hl * 8 + r;
        int col0 = n0 + wn + l16;
        int col1 = col0 + 16;
        size_t o0 = (size_t)row * N + col0;
        size_t o1 = (size_t)row * N + col1;
        Mout[o0] = (unsigned char)((F[o0] | (acc0[r] > 0)) ? 1 : 0);
        Mout[o1] = (unsigned char)((F[o1] | (acc1[r] > 0)) ? 1 : 0);
    }
}

__global__ __launch_bounds__(256)
void softmax_row_kernel(const float* __restrict__ S,
                        const float* __restrict__ adj,
                        const unsigned char* __restrict__ msk,
                        float* __restrict__ Out,
                        const float* __restrict__ Prev,
                        int N, int mode)
{
    const int r   = blockIdx.x;
    const int tid = threadIdx.x;
    float v[16];
    float mx = -3.4e38f;
    #pragma unroll
    for (int i = 0; i < 16; ++i) {
        int j = tid + (i << 8);
        size_t o = (size_t)r * N + j;
        float s = S[o];
        float x;
        if (mode == 0)      x = (adj[o] > 0.0f) ? s : -9.0e15f;
        else if (mode == 1) x = Prev[o] + ((adj[o] > 0.0f) ? s : -9.0e15f);
        else if (mode == 2) x = (adj[o] > 0.0f) ? -1.0e9f : s;
        else                x = msk[o] ? -1.0e9f : s;
        v[i] = x;
        mx = fmaxf(mx, x);
    }
    __shared__ float red[256];
    red[tid] = mx; __syncthreads();
    for (int s2 = 128; s2 > 0; s2 >>= 1) {
        if (tid < s2) red[tid] = fmaxf(red[tid], red[tid + s2]);
        __syncthreads();
    }
    mx = red[0]; __syncthreads();

    float sum = 0.0f;
    #pragma unroll
    for (int i = 0; i < 16; ++i) { v[i] = __expf(v[i] - mx); sum += v[i]; }
    red[tid] = sum; __syncthreads();
    for (int s2 = 128; s2 > 0; s2 >>= 1) {
        if (tid < s2) red[tid] += red[tid + s2];
        __syncthreads();
    }
    float inv = 1.0f / red[0];
    #pragma unroll
    for (int i = 0; i < 16; ++i)
        Out[(size_t)r * N + tid + (i << 8)] = v[i] * inv;
}

__global__ void bn_act_kernel(const float* __restrict__ H, const float* __restrict__ S,
                              const float* __restrict__ b, const float* __restrict__ g,
                              const float* __restrict__ bt, float* __restrict__ X,
                              size_t total, int C, int relu)
{
    size_t i = (size_t)blockIdx.x * blockDim.x + threadIdx.x;
    if (i >= total) return;
    int c = (int)(i % (size_t)C);
    float gs = g[c] * rsqrtf(1.0f + 1e-5f);
    float v = (H[i] + S[i] + b[c]) * gs + bt[c];
    X[i] = relu ? fmaxf(v, 0.0f) : v;
}

__global__ void add_bias_kernel(float* __restrict__ X, const float* __restrict__ b,
                                size_t total, int C)
{
    size_t i = (size_t)blockIdx.x * blockDim.x + threadIdx.x;
    if (i < total) X[i] += b[i % (size_t)C];
}

__global__ void add_kernel(const float* __restrict__ A, const float* __restrict__ B,
                           float* __restrict__ Z, size_t n)
{
    size_t i = (size_t)blockIdx.x * blockDim.x + threadIdx.x;
    if (i < n) Z[i] = A[i] + B[i];
}

__global__ void copy_kernel(const float* __restrict__ A, float* __restrict__ Z, size_t n)
{
    size_t i = (size_t)blockIdx.x * blockDim.x + threadIdx.x;
    if (i < n) Z[i] = A[i];
}

__global__ void make_flag_kernel(const float* __restrict__ adj,
                                 unsigned char* __restrict__ f, size_t n)
{
    size_t i = (size_t)blockIdx.x * blockDim.x + threadIdx.x;
    if (i < n) f[i] = (adj[i] > 0.0f) ? 1 : 0;
}

__global__ void floor_half_kernel(const float* __restrict__ A, float* __restrict__ Z,
                                  size_t n)
{
    size_t i = (size_t)blockIdx.x * blockDim.x + threadIdx.x;
    if (i < n) Z[i] = floorf(A[i] * 0.5f);
}

__global__ void mfb_norm_kernel(float* __restrict__ V, int C)
{
    int r = blockIdx.x, c = threadIdx.x;
    size_t o = (size_t)r * C + c;
    float x = V[o];
    float m = (x >= 0.0f) ? sqrtf(x) : -sqrtf(-x);
    __shared__ float red[128];
    red[c] = m * m; __syncthreads();
    for (int s = 64; s > 0; s >>= 1) {
        if (c < s) red[c] += red[c + s];
        __syncthreads();
    }
    float nrm = sqrtf(red[0]);
    V[o] = m / fmaxf(nrm, 1e-12f);
}

__global__ void logsoftmax_kernel(const float* __restrict__ X, float* __restrict__ Out,
                                  int C)
{
    int r = blockIdx.x, c = threadIdx.x;
    float x = (c < C) ? X[(size_t)r * C + c] : -3.4e38f;
    __shared__ float red[64];
    red[c] = x; __syncthreads();
    for (int s = 32; s > 0; s >>= 1) {
        if (c < s) red[c] = fmaxf(red[c], red[c + s]);
        __syncthreads();
    }
    float mx = red[0]; __syncthreads();
    float e = (c < C) ? __expf(x - mx) : 0.0f;
    red[c] = e; __syncthreads();
    for (int s = 32; s > 0; s >>= 1) {
        if (c < s) red[c] += red[c + s];
        __syncthreads();
    }
    if (c < C) Out[(size_t)r * C + c] = (x - mx) - __logf(red[0]);
}

extern "C" void kernel_launch(void* const* d_in, const int* in_sizes, int n_in,
                              void* d_out, int out_size, void* d_ws, size_t ws_size,
                              hipStream_t stream)
{
    const int N = NN, NF = NFD, NH = NHD, NC = NCD;
    (void)in_sizes; (void)n_in; (void)out_size; (void)ws_size;

    const float* fea = (const float*)d_in[0];
    const float* adj = (const float*)d_in[1];
    const float* ingc_w  = (const float*)d_in[2];
    const float* ingc_sw = (const float*)d_in[3];
    const float* ingc_b  = (const float*)d_in[4];
    const float* ingc_g  = (const float*)d_in[5];
    const float* ingc_bt = (const float*)d_in[6];
    const float* mu_w    = (const float*)d_in[7];
    const float* mu_sw   = (const float*)d_in[8];
    const float* mu_b    = (const float*)d_in[9];
    const float* mu_g    = (const float*)d_in[10];
    const float* mu_bt   = (const float*)d_in[11];
    const float* lv_w    = (const float*)d_in[12];
    const float* lv_sw   = (const float*)d_in[13];
    const float* lv_b    = (const float*)d_in[14];
    const float* lv_g    = (const float*)d_in[15];
    const float* lv_bt   = (const float*)d_in[16];
    const float* out_w   = (const float*)d_in[17];
    const float* out_sw  = (const float*)d_in[18];
    const float* out_b   = (const float*)d_in[19];
    const float* out_g   = (const float*)d_in[20];
    const float* out_bt  = (const float*)d_in[21];
    const float* key_w   = (const float*)d_in[22];
    const float* key_b   = (const float*)d_in[23];
    const float* q_w     = (const float*)d_in[24];
    const float* q_b     = (const float*)d_in[25];
    const float* mid_w   = (const float*)d_in[26];
    const float* mid_sw  = (const float*)d_in[27];
    const float* mid_b   = (const float*)d_in[28];
    const float* mid_g   = (const float*)d_in[29];
    const float* mid_bt  = (const float*)d_in[30];
    const float* mk_w    = (const float*)d_in[31];
    const float* mk_b    = (const float*)d_in[32];
    const float* mq_w    = (const float*)d_in[33];
    const float* mq_b    = (const float*)d_in[34];

    char* ws = (char*)d_ws;
    size_t off = 0;
    auto walloc = [&](size_t bytes) -> void* {
        void* p = ws + off;
        off += (bytes + 255) & ~(size_t)255;
        return p;
    };
    float*         S1     = (float*)walloc((size_t)N * N * sizeof(float));
    float*         ADJCON = (float*)walloc((size_t)N * N * sizeof(float));
    unsigned char* FLAG   = (unsigned char*)walloc((size_t)N * N);
    unsigned char* MASKP  = (unsigned char*)walloc((size_t)N * N);
    float*         Tb   = (float*)walloc((size_t)N * NH * sizeof(float));
    float*         Hb   = (float*)walloc((size_t)N * NH * sizeof(float));
    float*         Sb   = (float*)walloc((size_t)N * NH * sizeof(float));
    float*         Xb   = (float*)walloc((size_t)N * NH * sizeof(float));
    float*         MUb  = (float*)walloc((size_t)N * NH * sizeof(float));
    float*         LVb  = (float*)walloc((size_t)N * NH * sizeof(float));
    float*         KEYb = (float*)walloc((size_t)N * NH * sizeof(float));
    float*         QRYb = (float*)walloc((size_t)N * NH * sizeof(float));
    float*         VALb = (float*)walloc((size_t)N * NH * sizeof(float));
    float*         VINb = (float*)walloc((size_t)N * NH * sizeof(float));
    float*         OB   = (float*)walloc((size_t)N * NC * sizeof(float));

    float* out_con = (float*)d_out;
    float* out_mu  = out_con + (size_t)N * N;
    float* out_lv  = out_mu  + (size_t)N * NH;
    float* out_o   = out_lv  + (size_t)N * NH;

    const dim3 blk(256);
    const float inv_sqrt_nh = 0.08838834764831845f;

    auto gemm = [&](const float* A, int lda, const float* B, int ldb,
                    float* C, int ldc, int M, int Nc, int K,
                    float alpha, int beta, int transB) {
        dim3 grid((unsigned)((Nc + 63) / 64), (unsigned)((M + 63) / 64));
        if (transB)
            gemm_f16_wmma<1><<<grid, blk, 0, stream>>>(A, lda, B, ldb, C, ldc,
                                                       M, Nc, K, alpha, beta);
        else
            gemm_f16_wmma<0><<<grid, blk, 0, stream>>>(A, lda, B, ldb, C, ldc,
                                                       M, Nc, K, alpha, beta);
    };
    auto ew = [&](size_t n) { return dim3((unsigned)((n + 255) / 256)); };

    auto gcb = [&](const float* Xin, int Kin,
                   const float* w, const float* sw, const float* bb,
                   const float* gg, const float* btp,
                   float* Xo, int Cout, int relu) {
        gemm(Xin, Kin, w,  Cout, Tb, Cout, N, Cout, Kin, 1.0f, 0, 0);
        gemm(adj, N,   Tb, Cout, Hb, Cout, N, Cout, N,   1.0f, 0, 0);
        gemm(Xin, Kin, sw, Cout, Sb, Cout, N, Cout, Kin, 1.0f, 0, 0);
        size_t tot = (size_t)N * Cout;
        bn_act_kernel<<<ew(tot), blk, 0, stream>>>(Hb, Sb, bb, gg, btp, Xo,
                                                   tot, Cout, relu);
    };
    auto lin_cat = [&](const float* w, const float* bb, float* Out) {
        gemm(Xb,  NH, w,      NH + NF, Out, NH, N, NH, NH, 1.0f, 0, 1);
        gemm(fea, NF, w + NH, NH + NF, Out, NH, N, NH, NF, 1.0f, 1, 1);
        add_bias_kernel<<<ew((size_t)N * NH), blk, 0, stream>>>(Out, bb,
                                                 (size_t)N * NH, NH);
    };

    make_flag_kernel<<<ew((size_t)N * N), blk, 0, stream>>>(adj, FLAG, (size_t)N * N);

    gcb(fea, NF, ingc_w, ingc_sw, ingc_b, ingc_g, ingc_bt, Xb, NH, 1);
    gcb(Xb,  NH, mu_w,   mu_sw,   mu_b,   mu_g,   mu_bt,   MUb, NH, 1);
    gcb(Xb,  NH, lv_w,   lv_sw,   lv_b,   lv_g,   lv_bt,   LVb, NH, 1);

    gemm(MUb, NH, MUb, NH, S1, N, N, N, NH, 1.0f, 0, 1);
    softmax_row_kernel<<<N, blk, 0, stream>>>(S1, adj, nullptr, ADJCON, nullptr, N, 0);

    lin_cat(key_w, key_b, KEYb);
    gemm(Xb, NH, q_w, NH, QRYb, NH, N, NH, NH, 1.0f, 0, 1);
    add_bias_kernel<<<ew((size_t)N * NH), blk, 0, stream>>>(QRYb, q_b,
                                                            (size_t)N * NH, NH);
    gemm(KEYb, NH, QRYb, NH, S1, N, N, N, NH, inv_sqrt_nh, 0, 1);
    softmax_row_kernel<<<N, blk, 0, stream>>>(S1, adj, nullptr, S1, nullptr, N, 2);
    gemm(S1, N, KEYb, NH, VALb, NH, N, NH, N, 1.0f, 0, 0);
    add_kernel<<<ew((size_t)N * NH), blk, 0, stream>>>(VALb, Xb, VINb, (size_t)N * NH);

    adj2_mask_wmma<<<dim3(N / 64, N / 64), blk, 0, stream>>>(FLAG, MASKP, N);

    gcb(VINb, NH, mid_w, mid_sw, mid_b, mid_g, mid_bt, Xb, NH, 1);
    gcb(Xb,   NH, mu_w,  mu_sw,  mu_b,  mu_g,  mu_bt,  MUb, NH, 1);
    gcb(Xb,   NH, lv_w,  lv_sw,  lv_b,  lv_g,  lv_bt,  LVb, NH, 1);

    gemm(MUb, NH, MUb, NH, S1, N, N, N, NH, 1.0f, 0, 1);
    softmax_row_kernel<<<N, blk, 0, stream>>>(S1, adj, nullptr, ADJCON, ADJCON, N, 1);

    lin_cat(mk_w, mk_b, KEYb);
    gemm(Xb, NH, mq_w, NH, QRYb, NH, N, NH, NH, 1.0f, 0, 1);
    add_bias_kernel<<<ew((size_t)N * NH), blk, 0, stream>>>(QRYb, mq_b,
                                                            (size_t)N * NH, NH);
    gemm(KEYb, NH, QRYb, NH, S1, N, N, N, NH, inv_sqrt_nh, 0, 1);
    softmax_row_kernel<<<N, blk, 0, stream>>>(S1, nullptr, MASKP, S1, nullptr, N, 3);
    gemm(S1, N, KEYb, NH, VALb, NH, N, NH, N, 1.0f, 1, 0);

    mfb_norm_kernel<<<N, 128, 0, stream>>>(VALb, NH);
    add_kernel<<<ew((size_t)N * NH), blk, 0, stream>>>(VALb, Xb, VINb, (size_t)N * NH);

    gemm(VINb, NH, out_w,  NC, Tb, NC, N, NC, NH, 1.0f, 0, 0);
    gemm(adj,  N,  Tb,     NC, Hb, NC, N, NC, N,  1.0f, 0, 0);
    gemm(VINb, NH, out_sw, NC, Sb, NC, N, NC, NH, 1.0f, 0, 0);
    bn_act_kernel<<<ew((size_t)N * NC), blk, 0, stream>>>(Hb, Sb, out_b, out_g,
                                               out_bt, OB, (size_t)N * NC, NC, 0);
    logsoftmax_kernel<<<N, 64, 0, stream>>>(OB, out_o, NC);

    floor_half_kernel<<<ew((size_t)N * N), blk, 0, stream>>>(ADJCON, out_con,
                                                             (size_t)N * N);
    copy_kernel<<<ew((size_t)N * NH), blk, 0, stream>>>(MUb, out_mu, (size_t)N * NH);
    copy_kernel<<<ew((size_t)N * NH), blk, 0, stream>>>(LVb, out_lv, (size_t)N * NH);
}
